// Block_343597384085
// MI455X (gfx1250) — compile-verified, matrix operand order checked by reading, not run on hardware
//
#include <hip/hip_runtime.h>

#define D_MODEL 1024
#define SEQ     2048
#define BATCH   4
#define NSTEP   4
#define KCONV   4
#define EPS_F   1e-6f

#define NCHUNK  16
#define CHUNKL  (SEQ / NCHUNK)

typedef __attribute__((ext_vector_type(16))) __bf16 v16bf;
typedef __attribute__((ext_vector_type(8)))  float  v8f;
typedef int v4i_vs __attribute__((vector_size(16)));   // matches builtin param type

struct BF16FragBits { uint4 a, b; };   // 32 bytes == v16bf

#if __has_builtin(__builtin_amdgcn_global_load_async_to_lds_b128) && \
    __has_builtin(__builtin_amdgcn_s_wait_asynccnt)
#define HAVE_ASYNC_LDS 1
#else
#define HAVE_ASYNC_LDS 0
#endif

__device__ __forceinline__ unsigned short f2bf(float f) {
    union { float f; unsigned u; } v; v.f = f;
    unsigned r = v.u + 0x7FFFu + ((v.u >> 16) & 1u);   // round-to-nearest-even
    return (unsigned short)(r >> 16);
}
__device__ __forceinline__ float sigmoidf_(float x) { return 1.f / (1.f + __expf(-x)); }
__device__ __forceinline__ float siluf_(float x)    { return x / (1.f + __expf(-x)); }

// ---------------------------------------------------------------- elementwise
__global__ void cvt_bf16_kernel(const float* __restrict__ in,
                                unsigned short* __restrict__ out, int n) {
    int i = blockIdx.x * 256 + threadIdx.x;
    if (i < n) out[i] = f2bf(in[i]);
}

// fp32 [K][N] -> bf16 [N][K] (weight pre-transpose; 32x32 tiles, 32x8 threads)
__global__ __launch_bounds__(256)
void cvt_transpose_kernel(const float* __restrict__ in, unsigned short* __restrict__ out) {
    __shared__ float tile[32][33];
    const float* inp = in + (size_t)blockIdx.z * D_MODEL * D_MODEL;
    unsigned short* op = out + (size_t)blockIdx.z * D_MODEL * D_MODEL;
    int nb = blockIdx.x * 32;            // N base
    int kb = blockIdx.y * 32;            // K base
    int tx = threadIdx.x & 31, ty = threadIdx.x >> 5;
#pragma unroll
    for (int j = 0; j < 32; j += 8)
        tile[ty + j][tx] = inp[(size_t)(kb + ty + j) * D_MODEL + nb + tx];
    __syncthreads();
#pragma unroll
    for (int j = 0; j < 32; j += 8)
        op[(size_t)(nb + ty + j) * D_MODEL + kb + tx] = f2bf(tile[tx][ty + j]);
}

// causal depthwise conv (K=4) + SiLU; initializes c, ctx, out
__global__ void conv_kernel(const float* __restrict__ x,
                            const float* __restrict__ cw,
                            const float* __restrict__ cb,
                            float* __restrict__ c,
                            float* __restrict__ ctx,
                            float* __restrict__ out) {
    int i = blockIdx.x * 256 + threadIdx.x;          // B*S*D total, exact
    int d = i % D_MODEL;
    int s = (i / D_MODEL) % SEQ;
    int b = i / (D_MODEL * SEQ);
    float acc = cb[d];
#pragma unroll
    for (int k = 0; k < KCONV; k++) {
        int t = s - (KCONV - 1) + k;
        if (t >= 0) acc += x[((size_t)b * SEQ + t) * D_MODEL + d] * cw[k * D_MODEL + d];
    }
    float v = siluf_(acc);
    c[i] = v; ctx[i] = v; out[i] = v;
}

// u = c * betas * sqrt(clip(1 - alphas^2, 1e-6))
__global__ void u_kernel(const float* __restrict__ c,
                         const float* __restrict__ al,
                         const float* __restrict__ be,
                         float* __restrict__ u) {
    int i = blockIdx.x * 256 + threadIdx.x;
    float a = al[i];
    float ws = sqrtf(fmaxf(1.f - a * a, 1e-6f));
    u[i] = c[i] * be[i] * ws;
}

// ------------------------------------------------------------------ rmsnorm
__global__ __launch_bounds__(256)
void rms_kernel(const float* __restrict__ x, const float* __restrict__ w,
                unsigned short* __restrict__ out) {
    __shared__ float red[256];
    size_t base = (size_t)blockIdx.x * D_MODEL;
    float v[4]; float ss = 0.f;
#pragma unroll
    for (int j = 0; j < 4; j++) { v[j] = x[base + threadIdx.x + j * 256]; ss += v[j] * v[j]; }
    red[threadIdx.x] = ss; __syncthreads();
    for (int o = 128; o > 0; o >>= 1) {
        if ((int)threadIdx.x < o) red[threadIdx.x] += red[threadIdx.x + o];
        __syncthreads();
    }
    float scale = rsqrtf(red[0] * (1.f / D_MODEL) + EPS_F);
#pragma unroll
    for (int j = 0; j < 4; j++) {
        int d = threadIdx.x + j * 256;
        out[base + d] = f2bf(v[j] * scale * w[d]);
    }
}

// t = rmsnorm(out * gate) * fin_rms_w  -> bf16
__global__ __launch_bounds__(256)
void finalmix_kernel(const float* __restrict__ o, const float* __restrict__ g,
                     const float* __restrict__ w, unsigned short* __restrict__ tb) {
    __shared__ float red[256];
    size_t base = (size_t)blockIdx.x * D_MODEL;
    float v[4]; float ss = 0.f;
#pragma unroll
    for (int j = 0; j < 4; j++) {
        int d = threadIdx.x + j * 256;
        float t = o[base + d] * g[base + d];
        v[j] = t; ss += t * t;
    }
    red[threadIdx.x] = ss; __syncthreads();
    for (int oo = 128; oo > 0; oo >>= 1) {
        if ((int)threadIdx.x < oo) red[threadIdx.x] += red[threadIdx.x + oo];
        __syncthreads();
    }
    float scale = rsqrtf(red[0] * (1.f / D_MODEL) + EPS_F);
#pragma unroll
    for (int j = 0; j < 4; j++) {
        int d = threadIdx.x + j * 256;
        tb[base + d] = f2bf(v[j] * scale * w[d]);
    }
}

// ----------------------------------------------------------- chunked scan
__global__ void scan_part1(const float* __restrict__ al, const float* __restrict__ u,
                           float* __restrict__ cA, float* __restrict__ cH) {
    int i = blockIdx.x * 256 + threadIdx.x;          // B*NCHUNK*D
    int d  = i % D_MODEL;
    int ch = (i / D_MODEL) % NCHUNK;
    int b  = i / (D_MODEL * NCHUNK);
    size_t base = ((size_t)b * SEQ + (size_t)ch * CHUNKL) * D_MODEL + d;
    float h = 0.f, A = 1.f;
    for (int t = 0; t < CHUNKL; t++) {
        float a = al[base + (size_t)t * D_MODEL];
        h = a * h + u[base + (size_t)t * D_MODEL];
        A *= a;
    }
    cA[i] = A; cH[i] = h;
}

__global__ void scan_part2(const float* __restrict__ cA, const float* __restrict__ cH,
                           float* __restrict__ carry) {
    int i = blockIdx.x * 256 + threadIdx.x;          // B*D
    int d = i % D_MODEL;
    int b = i / D_MODEL;
    float h = 0.f;
    for (int ch = 0; ch < NCHUNK; ch++) {
        size_t idx = ((size_t)b * NCHUNK + ch) * D_MODEL + d;
        carry[idx] = h;
        h = cA[idx] * h + cH[idx];
    }
}

// re-walk with carry; fuse out += fetched and bf16(fetched)
__global__ void scan_part3(const float* __restrict__ al, const float* __restrict__ u,
                           const float* __restrict__ carry,
                           unsigned short* __restrict__ fb, float* __restrict__ out) {
    int i = blockIdx.x * 256 + threadIdx.x;          // B*NCHUNK*D
    int d  = i % D_MODEL;
    int ch = (i / D_MODEL) % NCHUNK;
    int b  = i / (D_MODEL * NCHUNK);
    size_t base = ((size_t)b * SEQ + (size_t)ch * CHUNKL) * D_MODEL + d;
    float h = carry[i];
    for (int t = 0; t < CHUNKL; t++) {
        size_t idx = base + (size_t)t * D_MODEL;
        h = al[idx] * h + u[idx];
        fb[idx] = f2bf(h);
        out[idx] += h;
    }
}

// ------------------------------------------------------------- WMMA GEMM
// C[M,N] = act(A[M,K](bf16) @ Bt[N,K](bf16, pre-transposed) + bias), opt +=.
// Block tile 128x128, BK=64 (two WMMA k-steps per stage), 8 waves 4(M)x2(N),
// each wave 2x4 v_wmma_f32_16x16x32_bf16 accumulators. Tiles staged via
// async global->LDS (ASYNCcnt) when the toolchain exposes the builtin.
#define BM 128
#define BN 128
#define BK 64
#define LDK 72   // ushort stride; 144 bytes, 16B-aligned rows

template <int ACT, bool ADD>
__global__ __launch_bounds__(256)
void gemm_bf16(const unsigned short* __restrict__ A,
               const unsigned short* __restrict__ Bt,
               const float* __restrict__ bias,
               float* __restrict__ C, int M, int N, int K) {
    __shared__ __align__(16) unsigned short sA[BM][LDK];
    __shared__ __align__(16) unsigned short sB[BN][LDK];   // sB[n][k]

    const int tid   = threadIdx.x;
    const int lane  = tid & 31;
    const int wave  = tid >> 5;
    const int waveM = wave >> 1;       // 0..3
    const int waveN = wave & 1;        // 0..1
    const int tileM = blockIdx.y * BM;
    const int tileN = blockIdx.x * BN;
    const int m16   = lane & 15;
    const int hh    = lane >> 4;       // half-wave K-group select (ISA layout)

    v8f acc[2][4] = {};

    for (int k0 = 0; k0 < K; k0 += BK) {
        __syncthreads();
        // stage A[128][64] and Bt[128][64]: 1024 16B chunks each, 4/thread
#pragma unroll
        for (int cch = 0; cch < 4; cch++) {
            int chunk = tid + cch * 256;
            int row = chunk >> 3;            // 8 chunks per 64-wide row
            int ks  = (chunk & 7) << 3;
            const unsigned short* gA = A  + (size_t)(tileM + row) * K + k0 + ks;
            const unsigned short* gB = Bt + (size_t)(tileN + row) * K + k0 + ks;
#if HAVE_ASYNC_LDS
            __builtin_amdgcn_global_load_async_to_lds_b128(
                (__attribute__((address_space(1))) v4i_vs*)gA,
                (__attribute__((address_space(3))) v4i_vs*)&sA[row][ks], 0, 0);
            __builtin_amdgcn_global_load_async_to_lds_b128(
                (__attribute__((address_space(1))) v4i_vs*)gB,
                (__attribute__((address_space(3))) v4i_vs*)&sB[row][ks], 0, 0);
#else
            *(uint4*)&sA[row][ks] = *(const uint4*)gA;
            *(uint4*)&sB[row][ks] = *(const uint4*)gB;
#endif
        }
#if HAVE_ASYNC_LDS
        __builtin_amdgcn_s_wait_asynccnt(0);
#endif
        __syncthreads();

#pragma unroll
        for (int kb = 0; kb < BK; kb += 32) {
            v16bf af[2], bfr[4];
#pragma unroll
            for (int mi = 0; mi < 2; mi++) {
                const unsigned short* rp = &sA[waveM * 32 + mi * 16 + m16][kb];
                BF16FragBits t;
                t.a = *(const uint4*)(rp + 8 * hh);        // K = 8h .. 8h+7
                t.b = *(const uint4*)(rp + 16 + 8 * hh);   // K = 16+8h ..
                af[mi] = __builtin_bit_cast(v16bf, t);
            }
#pragma unroll
            for (int ni = 0; ni < 4; ni++) {
                const unsigned short* rp = &sB[waveN * 64 + ni * 16 + m16][kb];
                BF16FragBits t;
                t.a = *(const uint4*)(rp + 8 * hh);
                t.b = *(const uint4*)(rp + 16 + 8 * hh);
                bfr[ni] = __builtin_bit_cast(v16bf, t);
            }
#pragma unroll
            for (int mi = 0; mi < 2; mi++)
#pragma unroll
                for (int ni = 0; ni < 4; ni++)
                    acc[mi][ni] = __builtin_amdgcn_wmma_f32_16x16x32_bf16(
                        false, af[mi], false, bfr[ni], (short)0, acc[mi][ni],
                        false, false);
        }
    }

    // epilogue: VGPR r of a 16x16 f32 D-tile holds row (r + 8*hh), col = m16
#pragma unroll
    for (int mi = 0; mi < 2; mi++) {
#pragma unroll
        for (int ni = 0; ni < 4; ni++) {
            int col = tileN + waveN * 64 + ni * 16 + m16;
            float bv = bias ? bias[col] : 0.f;
            int rowBase = tileM + waveM * 32 + mi * 16 + 8 * hh;
#pragma unroll
            for (int r = 0; r < 8; r++) {
                float val = acc[mi][ni][r] + bv;
                if (ACT == 1) val = sigmoidf_(val);
                else if (ACT == 2) val = siluf_(val);
                size_t oidx = (size_t)(rowBase + r) * N + col;
                if (ADD) C[oidx] += val;
                else     C[oidx] = val;
            }
        }
    }
}

// ---------------------------------------------------------------- launcher
extern "C" void kernel_launch(void* const* d_in, const int* in_sizes, int n_in,
                              void* d_out, int out_size, void* d_ws, size_t ws_size,
                              hipStream_t stream) {
    const float* x        = (const float*)d_in[0];
    const float* conv_w   = (const float*)d_in[1];
    const float* conv_b   = (const float*)d_in[2];
    const float* gate_w   = (const float*)d_in[3];
    const float* gate_b   = (const float*)d_in[4];
    const float* rms_w    = (const float*)d_in[5];
    const float* alpha_w  = (const float*)d_in[6];
    const float* alpha_b  = (const float*)d_in[7];
    const float* beta_w   = (const float*)d_in[8];
    const float* beta_b   = (const float*)d_in[9];
    const float* ctx_w    = (const float*)d_in[10];
    const float* ctx_b    = (const float*)d_in[11];
    const float* fin_rmsw = (const float*)d_in[12];
    const float* fin_w    = (const float*)d_in[13];
    const float* fin_b    = (const float*)d_in[14];
    float* outp = (float*)d_out;

    const size_t BSD = (size_t)BATCH * SEQ * D_MODEL;
    const size_t DD  = (size_t)D_MODEL * D_MODEL;

    char* p = (char*)d_ws;
    auto alloc = [&](size_t bytes) -> void* {
        void* r = (void*)p;
        p += (bytes + 255) & ~(size_t)255;
        return r;
    };

    unsigned short* xb  = (unsigned short*)alloc(BSD * 2);
    unsigned short* wg  = (unsigned short*)alloc(DD * 2);              // transposed
    unsigned short* wa  = (unsigned short*)alloc((size_t)NSTEP * DD * 2);
    unsigned short* wb  = (unsigned short*)alloc((size_t)NSTEP * DD * 2);
    unsigned short* wc  = (unsigned short*)alloc((size_t)NSTEP * DD * 2);
    unsigned short* wf  = (unsigned short*)alloc(DD * 2);
    float* c      = (float*)alloc(BSD * 4);
    float* ctx    = (float*)alloc(BSD * 4);
    float* out    = (float*)alloc(BSD * 4);
    float* gate   = (float*)alloc(BSD * 4);
    float* alphas = (float*)alloc(BSD * 4);
    float* betas  = (float*)alloc(BSD * 4);
    float* u      = (float*)alloc(BSD * 4);
    unsigned short* cnb = (unsigned short*)alloc(BSD * 2);
    unsigned short* fb  = (unsigned short*)alloc(BSD * 2);
    unsigned short* tb  = (unsigned short*)alloc(BSD * 2);
    float* cA    = (float*)alloc((size_t)BATCH * NCHUNK * D_MODEL * 4);
    float* cH    = (float*)alloc((size_t)BATCH * NCHUNK * D_MODEL * 4);
    float* carry = (float*)alloc((size_t)BATCH * NCHUNK * D_MODEL * 4);

    const int ROWS = BATCH * SEQ;

    // input -> bf16; weights -> transposed bf16
    cvt_bf16_kernel<<<(unsigned)((BSD + 255) / 256), 256, 0, stream>>>(x, xb, (int)BSD);
    dim3 tgrid(D_MODEL / 32, D_MODEL / 32, 1);
    dim3 tgridN(D_MODEL / 32, D_MODEL / 32, NSTEP);
    cvt_transpose_kernel<<<tgrid,  256, 0, stream>>>(gate_w,  wg);
    cvt_transpose_kernel<<<tgridN, 256, 0, stream>>>(alpha_w, wa);
    cvt_transpose_kernel<<<tgridN, 256, 0, stream>>>(beta_w,  wb);
    cvt_transpose_kernel<<<tgridN, 256, 0, stream>>>(ctx_w,   wc);
    cvt_transpose_kernel<<<tgrid,  256, 0, stream>>>(fin_w,   wf);

    conv_kernel<<<(unsigned)(BSD / 256), 256, 0, stream>>>(x, conv_w, conv_b, c, ctx, out);

    dim3 ggrid(D_MODEL / BN, ROWS / BM);
    gemm_bf16<2, false><<<ggrid, 256, 0, stream>>>(xb, wg, gate_b, gate, ROWS, D_MODEL, D_MODEL);

    for (int i = 0; i < NSTEP; i++) {
        rms_kernel<<<ROWS, 256, 0, stream>>>(ctx, rms_w + (size_t)i * D_MODEL, cnb);
        gemm_bf16<1, false><<<ggrid, 256, 0, stream>>>(cnb, wa + (size_t)i * DD,
            alpha_b + (size_t)i * D_MODEL, alphas, ROWS, D_MODEL, D_MODEL);
        gemm_bf16<2, false><<<ggrid, 256, 0, stream>>>(cnb, wb + (size_t)i * DD,
            beta_b + (size_t)i * D_MODEL, betas, ROWS, D_MODEL, D_MODEL);
        u_kernel<<<(unsigned)(BSD / 256), 256, 0, stream>>>(c, alphas, betas, u);
        scan_part1<<<(unsigned)((size_t)BATCH * NCHUNK * D_MODEL / 256), 256, 0, stream>>>(alphas, u, cA, cH);
        scan_part2<<<(unsigned)((size_t)BATCH * D_MODEL / 256), 256, 0, stream>>>(cA, cH, carry);
        scan_part3<<<(unsigned)((size_t)BATCH * NCHUNK * D_MODEL / 256), 256, 0, stream>>>(alphas, u, carry, fb, out);
        gemm_bf16<2, true><<<ggrid, 256, 0, stream>>>(fb, wc + (size_t)i * DD,
            ctx_b + (size_t)i * D_MODEL, ctx, ROWS, D_MODEL, D_MODEL);
    }

    finalmix_kernel<<<ROWS, 256, 0, stream>>>(out, gate, fin_rmsw, tb);
    gemm_bf16<2, false><<<ggrid, 256, 0, stream>>>(tb, wf, fin_b, outp, ROWS, D_MODEL, D_MODEL);
}
